// PAM_Module_Dep_52055003628044
// MI455X (gfx1250) — matrix operand order repaired and checked by reading, not compiled, not run on hardware
//
#include <hip/hip_runtime.h>
#include <hip/hip_bf16.h>

// ---------------------------------------------------------------------------
// PAM (position attention) with depth, flash-attention style, WMMA f16.
//   B=4, C=512, H=W=64 -> N=4096, D=128 (concat q/k dim), CQ=64.
// Workspace layout (f16):
//   Qh [B][N][128]  (4 MiB)   row-major per query position (A/B frags for S)
//   Kh [B][N][128]  (4 MiB)
//   Vh [B][C][N]    (16 MiB)  TRANSPOSED: A-frags for O^T load contiguous
// ---------------------------------------------------------------------------

typedef _Float16 half_t;
typedef __attribute__((ext_vector_type(16))) _Float16     v16h;
typedef __attribute__((ext_vector_type(4)))  _Float16     v4h;
typedef __attribute__((ext_vector_type(8)))  float        v8f;
typedef __attribute__((ext_vector_type(4)))  float        v4f;
typedef __attribute__((ext_vector_type(4)))  unsigned int v4u;

#define BATCH 4
#define CCH   512
#define NPOS  4096
#define DQK   128

__device__ __forceinline__ v8f wmma_f16(v16h a, v16h b, v8f c) {
  // D = A(16x32 f16) * B(32x16 f16) + C(16x16 f32)
  return __builtin_amdgcn_wmma_f32_16x16x32_f16(
      /*neg_a=*/false, a, /*neg_b=*/false, b,
      /*c_mod=*/(short)0, c, /*reuse_a=*/false, /*reuse_b=*/false);
}

// ---------------------------------------------------------------------------
// Projection GEMM: out = (f16)( W[d,:] . src[:,n] + bias[d] )
//   src : [C][N] f32 per batch (batch = blockIdx.z)
//   W   : [Dlaunch][C] f32,  bias : [Dlaunch] f32
//   transposed==0 : out[(b*N + n)*ldo + doff + dlocal]   (Q/K, ldo=128)
//   transposed==1 : out[((b*Dtot + d)*N) + n], Dtot=ldo  (V, contiguous in n)
// Block: 256 threads (8 waves). Tile: 64 n x 64 d, K-steps of 32 over C=512.
// ---------------------------------------------------------------------------
__global__ void proj_gemm_kernel(const float* __restrict__ src,
                                 const float* __restrict__ W,
                                 const float* __restrict__ bias,
                                 half_t* __restrict__ outp,
                                 int doff, int ldo, int transposed) {
  __shared__ __align__(16) half_t Xs[64 * 32];  // [m(n-dim)][k(c-dim)]
  __shared__ __align__(16) half_t Ws[64 * 32];  // [n(d-dim)][k(c-dim)]

  const int tid  = threadIdx.x;
  const int lane = tid & 31;
  const int w    = tid >> 5;
  const int col  = lane & 15;
  const int hf   = lane >> 4;

  const int n0    = blockIdx.x * 64;
  const int dbase = blockIdx.y * 64;
  const int b     = blockIdx.z;

  const int mt  = w >> 1;         // n-tile (0..3)
  const int ntb = (w & 1) * 2;    // d-tile pair base (0 or 2)

  v8f acc0 = {};
  v8f acc1 = {};

  const size_t srcb = (size_t)b * CCH * NPOS;
  const int ms = tid & 63;          // staging row (n for X / d for W)
  const int kg = (tid >> 6) * 4;    // staging k group base

  for (int c0 = 0; c0 < CCH; c0 += 32) {
    // --- stage X tile (64 n x 32 c): 4 coalesced b32 loads -> ds_store_b64 ---
#pragma unroll
    for (int g = 0; g < 2; ++g) {
      int kb = kg + g * 16;
      const float* sp = src + srcb + (size_t)(c0 + kb) * NPOS + n0 + ms;
      v4f xv;
      xv[0] = sp[0 * NPOS];
      xv[1] = sp[1 * NPOS];
      xv[2] = sp[2 * NPOS];
      xv[3] = sp[3 * NPOS];
      *(v4h*)(Xs + ms * 32 + kb) = __builtin_convertvector(xv, v4h);
    }
    // --- stage W tile (64 d x 32 c): one b128 load -> ds_store_b64 ---
#pragma unroll
    for (int g = 0; g < 2; ++g) {
      int kb = kg + g * 16;
      v4f wv = *(const v4f*)(W + (size_t)(dbase + ms) * CCH + c0 + kb);
      *(v4h*)(Ws + ms * 32 + kb) = __builtin_convertvector(wv, v4h);
    }
    __syncthreads();

    // A fragment: A[m,k] = Xs[mt*16+m][k]  (two 8-half runs per lane: k from 8*hf and from 16+8*hf)
    v16h a;
#pragma unroll
    for (int v = 0; v < 8; ++v) {
      int k = ((v >> 2) << 4) + (hf << 3) + ((v & 3) << 1);
      a[2 * v]     = Xs[(mt * 16 + col) * 32 + k];
      a[2 * v + 1] = Xs[(mt * 16 + col) * 32 + k + 1];
    }
#pragma unroll
    for (int t = 0; t < 2; ++t) {
      int nt = ntb + t;
      v16h bf;
#pragma unroll
      for (int v = 0; v < 8; ++v) {
        int k = ((v >> 2) << 4) + (hf << 3) + ((v & 3) << 1);
        bf[2 * v]     = Ws[(nt * 16 + col) * 32 + k];
        bf[2 * v + 1] = Ws[(nt * 16 + col) * 32 + k + 1];
      }
      if (t == 0) acc0 = wmma_f16(a, bf, acc0);
      else        acc1 = wmma_f16(a, bf, acc1);
    }
    __syncthreads();
  }

  // epilogue: bias + convert + store f16
#pragma unroll
  for (int t = 0; t < 2; ++t) {
    int nt = ntb + t;
    int dl = nt * 16 + col;                 // d local to this launch's dbase
    float bv = bias[dbase + dl];
    if (transposed) {
      // out[(b*Dtot + dbase+dl)*N + n]: consecutive r -> consecutive n
      half_t* op = outp + ((size_t)b * ldo + dbase + dl) * NPOS + n0 + mt * 16 + (hf << 3);
#pragma unroll
      for (int r = 0; r < 8; ++r)
        op[r] = (half_t)((t == 0 ? acc0[r] : acc1[r]) + bv);
    } else {
#pragma unroll
      for (int r = 0; r < 8; ++r) {
        int ng = n0 + mt * 16 + r + (hf << 3);
        outp[((size_t)b * NPOS + ng) * ldo + doff + dl] =
            (half_t)((t == 0 ? acc0[r] : acc1[r]) + bv);
      }
    }
  }
}

// ---------------------------------------------------------------------------
// Flash attention: per workgroup one (batch, 32-query tile).
//  - S(32x64) = Q * K^T via WMMA, staged to LDS
//  - online softmax stats (wave0, one row per lane)
//  - O^T(512x32) += V * P^T via WMMA (each wave owns 64 channels)
//  - finalize: out = gamma * O/l + x   (coalesced along i)
// ---------------------------------------------------------------------------
__global__ void flash_attn_kernel(const half_t* __restrict__ Qh,
                                  const half_t* __restrict__ Kh,
                                  const half_t* __restrict__ Vh,
                                  const float*  __restrict__ x,
                                  const float*  __restrict__ gamma,
                                  float* __restrict__ outp) {
  __shared__ __align__(16) half_t Qs[32 * 128];  // [m(i)][k(d)] pairs contiguous
  __shared__ float  Ss[32 * 65];                 // padded stride 65 (bank-safe)
  __shared__ __align__(16) half_t Ps[32 * 66];   // padded stride 66, 4B-aligned pairs
  __shared__ float  mrow[32], lrow[32], srow[32];

  const int tid  = threadIdx.x;
  const int lane = tid & 31;
  const int w    = tid >> 5;
  const int col  = lane & 15;
  const int hf   = lane >> 4;

  const int i0 = blockIdx.x * 32;
  const int b  = blockIdx.y;

  // stage Q tile: 8 KiB contiguous, b128 copies
  {
    const v4u* qs = (const v4u*)(Qh + ((size_t)b * NPOS + i0) * DQK);
    v4u* qd = (v4u*)Qs;
#pragma unroll
    for (int r = 0; r < 2; ++r) qd[tid + r * 256] = qs[tid + r * 256];
  }
  if (tid < 32) { mrow[tid] = -1e30f; lrow[tid] = 0.0f; }
  __syncthreads();

  v8f o[2][4];  // [it: query sub-tile][ct: 16-channel tile], rows = channels
  {
    v8f z = {};
#pragma unroll
    for (int it = 0; it < 2; ++it)
#pragma unroll
      for (int ct = 0; ct < 4; ++ct) o[it][ct] = z;
  }

  const int rtS = w >> 2;  // S row-tile owned by this wave
  const int jtS = w & 3;   // S col-tile owned by this wave

  for (int j0 = 0; j0 < NPOS; j0 += 64) {
    if (lane == 0 && (j0 + 64) < NPOS)  // prefetch next K tile into L2
      __builtin_prefetch((const void*)(Kh + ((size_t)b * NPOS + j0 + 64) * DQK), 0, 1);

    // ---- 1. S tile: 16x16 per wave, K-dim = 128 in 4 WMMA steps ----
    {
      v8f s = {};
#pragma unroll
      for (int kk = 0; kk < 4; ++kk) {
        v16h a, bf;
#pragma unroll
        for (int v = 0; v < 8; ++v) {
          int k = kk * 32 + ((v >> 2) << 4) + (hf << 3) + ((v & 3) << 1);
          a[2 * v]     = Qs[(rtS * 16 + col) * DQK + k];
          a[2 * v + 1] = Qs[(rtS * 16 + col) * DQK + k + 1];
        }
        // B[k=d, n=j] = K[j0+jt*16+n][kk*32+k]: 2x16B contiguous per lane
        const half_t* kp = Kh + ((size_t)b * NPOS + j0 + jtS * 16 + col) * DQK + kk * 32;
#pragma unroll
        for (int v = 0; v < 8; ++v) {
          int k = ((v >> 2) << 4) + (hf << 3) + ((v & 3) << 1);
          bf[2 * v]     = kp[k];
          bf[2 * v + 1] = kp[k + 1];
        }
        s = wmma_f16(a, bf, s);
      }
#pragma unroll
      for (int r = 0; r < 8; ++r)
        Ss[(rtS * 16 + r + hf * 8) * 65 + jtS * 16 + col] = s[r];
    }
    __syncthreads();

    // ---- 2. online softmax stats: one row per lane of wave 0 ----
    if (tid < 32) {
      int i = tid;
      float mold = mrow[i], mnew = mold;
#pragma unroll 8
      for (int j = 0; j < 64; ++j) mnew = fmaxf(mnew, Ss[i * 65 + j]);
      float sc = __expf(mold - mnew);
      float sum = 0.0f;
#pragma unroll 8
      for (int j = 0; j < 64; ++j) {
        float p = __expf(Ss[i * 65 + j] - mnew);
        Ps[i * 66 + j] = (half_t)p;
        sum += p;
      }
      mrow[i] = mnew;
      lrow[i] = lrow[i] * sc + sum;
      srow[i] = sc;
    }
    __syncthreads();

    // ---- 3. rescale O, then O^T += V * P^T ----
#pragma unroll
    for (int it = 0; it < 2; ++it) {
      float sc = srow[it * 16 + col];  // per-lane: column = query index
#pragma unroll
      for (int ct = 0; ct < 4; ++ct) o[it][ct] *= sc;
    }
#pragma unroll
    for (int jj = 0; jj < 2; ++jj) {
      // B fragments: B[k=j, n=i] = P[i, jj*32+k] (contiguous pairs in LDS)
      v16h bfr0, bfr1;
#pragma unroll
      for (int v = 0; v < 8; ++v) {
        int k = jj * 32 + ((v >> 2) << 4) + (hf << 3) + ((v & 3) << 1);
        bfr0[2 * v]     = Ps[(0 * 16 + col) * 66 + k];
        bfr0[2 * v + 1] = Ps[(0 * 16 + col) * 66 + k + 1];
        bfr1[2 * v]     = Ps[(1 * 16 + col) * 66 + k];
        bfr1[2 * v + 1] = Ps[(1 * 16 + col) * 66 + k + 1];
      }
#pragma unroll
      for (int ct = 0; ct < 4; ++ct) {
        // A[m=c, k=j] = V[b, cbase+m, j0+jj*32+k]: Vh is [B][C][N] so each
        // lane reads 2x16B contiguous runs -> global_load_b128
        v16h a;
        const half_t* vp =
            Vh + ((size_t)b * CCH + w * 64 + ct * 16 + col) * NPOS + j0 + jj * 32;
#pragma unroll
        for (int v = 0; v < 8; ++v) {
          int k = ((v >> 2) << 4) + (hf << 3) + ((v & 3) << 1);
          a[2 * v]     = vp[k];
          a[2 * v + 1] = vp[k + 1];
        }
        o[0][ct] = wmma_f16(a, bfr0, o[0][ct]);
        o[1][ct] = wmma_f16(a, bfr1, o[1][ct]);
      }
    }
    __syncthreads();
  }

  // ---- finalize: out[b,c,i] = gamma * O[c,i]/l[i] + x[b,c,i] ----
  const float g = gamma[0];
#pragma unroll
  for (int it = 0; it < 2; ++it) {
    int i = i0 + it * 16 + col;
    float invl = 1.0f / lrow[it * 16 + col];
#pragma unroll
    for (int ct = 0; ct < 4; ++ct) {
#pragma unroll
      for (int r = 0; r < 8; ++r) {
        int c = w * 64 + ct * 16 + r + hf * 8;
        size_t idx = ((size_t)b * CCH + c) * NPOS + i;  // 16 consecutive i / half-wave
        outp[idx] = g * o[it][ct][r] * invl + x[idx];
      }
    }
  }
}

// ---------------------------------------------------------------------------
extern "C" void kernel_launch(void* const* d_in, const int* in_sizes, int n_in,
                              void* d_out, int out_size, void* d_ws, size_t ws_size,
                              hipStream_t stream) {
  const float* x     = (const float*)d_in[0];
  const float* dep   = (const float*)d_in[1];
  const float* wq    = (const float*)d_in[2];
  const float* bq    = (const float*)d_in[3];
  const float* wqd   = (const float*)d_in[4];
  const float* bqd   = (const float*)d_in[5];
  const float* wk    = (const float*)d_in[6];
  const float* bk    = (const float*)d_in[7];
  const float* wkd   = (const float*)d_in[8];
  const float* bkd   = (const float*)d_in[9];
  const float* wv    = (const float*)d_in[10];
  const float* bv    = (const float*)d_in[11];
  const float* gamma = (const float*)d_in[12];
  float* outp = (float*)d_out;

  half_t* Qh = (half_t*)d_ws;
  half_t* Kh = Qh + (size_t)BATCH * NPOS * DQK;
  half_t* Vh = Kh + (size_t)BATCH * NPOS * DQK;

  dim3 blk(256);
  // Q/K projections: concat(x-part, dep-part) -> [N][128] f16
  proj_gemm_kernel<<<dim3(NPOS / 64, 1, BATCH), blk, 0, stream>>>(x,   wq,  bq,  Qh, 0,  DQK, 0);
  proj_gemm_kernel<<<dim3(NPOS / 64, 1, BATCH), blk, 0, stream>>>(dep, wqd, bqd, Qh, 64, DQK, 0);
  proj_gemm_kernel<<<dim3(NPOS / 64, 1, BATCH), blk, 0, stream>>>(x,   wk,  bk,  Kh, 0,  DQK, 0);
  proj_gemm_kernel<<<dim3(NPOS / 64, 1, BATCH), blk, 0, stream>>>(dep, wkd, bkd, Kh, 64, DQK, 0);
  // V projection -> [C][N] f16 (transposed layout for contiguous A-frags)
  proj_gemm_kernel<<<dim3(NPOS / 64, CCH / 64, BATCH), blk, 0, stream>>>(x, wv, bv, Vh, 0, CCH, 1);
  // fused flash attention + gamma*out + x
  flash_attn_kernel<<<dim3(NPOS / 32, BATCH), blk, 0, stream>>>(Qh, Kh, Vh, x, gamma, outp);
}
